// AttentionPooling_10591389352018
// MI455X (gfx1250) — matrix operand order repaired and checked by reading, not compiled, not run on hardware
//
#include <hip/hip_runtime.h>
#include <hip/hip_bf16.h>

typedef __attribute__((ext_vector_type(16))) __bf16 v16bf;
typedef __attribute__((ext_vector_type(8)))  float  v8f;

#define DIM  128
#define NSEG 16384

// ---- monotonic float<->uint ordinal (for atomicMax on floats of any sign) ----
__device__ __forceinline__ unsigned f2ord(float f) {
  unsigned u = __float_as_uint(f);
  return (u & 0x80000000u) ? ~u : (u | 0x80000000u);
}
__device__ __forceinline__ float ord2f(unsigned u) {
  return (u & 0x80000000u) ? __uint_as_float(u & 0x7FFFFFFFu)
                           : __uint_as_float(~u);
}

// ---------------- Pass 1: gate = x@Wg + bg ; segment max (ordinal atomicMax) --
__global__ __launch_bounds__(256)
void gate_pass(const float* __restrict__ x, const int* __restrict__ index,
               const float* __restrict__ Wg, const float* __restrict__ bg,
               float* __restrict__ gate_raw, unsigned* __restrict__ segmax,
               int N) {
  const int lane   = threadIdx.x & 31;
  const int wave   = (blockIdx.x * blockDim.x + threadIdx.x) >> 5;
  const int nwaves = (gridDim.x * blockDim.x) >> 5;
  const float4 wg  = reinterpret_cast<const float4*>(Wg)[lane];
  const float  bgv = bg[0];
  const float4* x4 = reinterpret_cast<const float4*>(x);
  for (int row = wave; row < N; row += nwaves) {
    float4 xv = x4[(size_t)row * (DIM / 4) + lane];
    float p = xv.x * wg.x + xv.y * wg.y + xv.z * wg.z + xv.w * wg.w;
    #pragma unroll
    for (int o = 16; o > 0; o >>= 1) p += __shfl_xor(p, o, 32);
    if (lane == 0) {
      float g = p + bgv;
      gate_raw[row] = g;
      atomicMax(&segmax[index[row]], f2ord(g));
    }
  }
}

// ---------------- Pass 2: e = exp(g - segmax) ; denom += e -------------------
__global__ __launch_bounds__(256)
void exp_pass(float* __restrict__ gate_raw, const int* __restrict__ index,
              const unsigned* __restrict__ segmax, float* __restrict__ denom,
              int N) {
  int n = blockIdx.x * blockDim.x + threadIdx.x;
  if (n >= N) return;
  int s = index[n];
  float e = __expf(gate_raw[n] - ord2f(segmax[s]));
  gate_raw[n] = e;                 // reuse buffer: now holds e
  atomicAdd(&denom[s], e);
}

// ---------------- Pass 3: per-segment gate mass ------------------------------
__global__ __launch_bounds__(256)
void gsum_pass(const float* __restrict__ denom, float* __restrict__ gsum, int S) {
  int s = blockIdx.x * blockDim.x + threadIdx.x;
  if (s >= S) return;
  float d = denom[s];
  gsum[s] = d / (d + 1e-10f);      // == segment_sum(gate); 0 for empty segments
}

// ---------------- Pass 3b: pre-pack Wm into WMMA B-lane bf16 layout ----------
// Bpack layout: [tileN(8)][t(4)][lane(32)][16 bf16]  (32 KB total, launch-const)
__global__ __launch_bounds__(256)
void pack_B(const float* __restrict__ Wm, __bf16* __restrict__ Bpack) {
  int id = blockIdx.x * blockDim.x + threadIdx.x;   // 0 .. 1023
  if (id >= (DIM / 16) * 4 * 32) return;
  int lane  = id & 31;
  int t     = (id >> 5) & 3;
  int tileN = id >> 7;
  int l15 = lane & 15;
  int hi  = lane >> 4;
  // B (32x16 bf16, ISA layout): lane -> N=l15; element j -> K=(j<8 ? j : j+8) (+8 if hi-half)
  const float* bcol = Wm + (size_t)(t * 32 + (hi ? 8 : 0)) * DIM + tileN * 16 + l15;
  __bf16* dst = Bpack + (size_t)id * 16;
  #pragma unroll
  for (int j = 0; j < 16; ++j) dst[j] = (__bf16)bcol[(size_t)(j < 8 ? j : j + 8) * DIM];
}

// ---------------- Pass 4: y[s] += w_n * x[n]  (sorted index, boundary flush) -
// Chunks assigned in REVERSE order so this pass starts on the L2-hot tail of x
// left behind by gate_pass (x = 256 MB vs 192 MB L2).
__global__ __launch_bounds__(256)
void segsum_pass(const float* __restrict__ x, const int* __restrict__ index,
                 const float* __restrict__ e, const float* __restrict__ denom,
                 float* __restrict__ y, int N, int numWaves) {
  const int lane = threadIdx.x & 31;
  const int wave = (blockIdx.x * blockDim.x + threadIdx.x) >> 5;
  if (wave >= numWaves) return;
  const int chunk = numWaves - 1 - wave;            // reverse assignment
  const int r0 = chunk * 128;
  const int r1 = (r0 + 128 < N) ? (r0 + 128) : N;
  const float4* x4 = reinterpret_cast<const float4*>(x);
  float4 acc = make_float4(0.f, 0.f, 0.f, 0.f);
  int cur = -1;
  for (int r = r0; r < r1; ++r) {
    int s = index[r];                               // uniform across the wave
    if (s != cur) {
      if (cur >= 0) {
        float* yr = y + (size_t)cur * DIM + lane * 4;
        atomicAdd(yr + 0, acc.x); atomicAdd(yr + 1, acc.y);
        atomicAdd(yr + 2, acc.z); atomicAdd(yr + 3, acc.w);
      }
      cur = s;
      acc = make_float4(0.f, 0.f, 0.f, 0.f);
    }
    float w = e[r] / (denom[s] + 1e-10f);
    float4 xv = x4[(size_t)r * (DIM / 4) + lane];
    acc.x += w * xv.x; acc.y += w * xv.y; acc.z += w * xv.z; acc.w += w * xv.w;
  }
  if (cur >= 0) {
    float* yr = y + (size_t)cur * DIM + lane * 4;
    atomicAdd(yr + 0, acc.x); atomicAdd(yr + 1, acc.y);
    atomicAdd(yr + 2, acc.z); atomicAdd(yr + 3, acc.w);
  }
}

// ---------------- Pass 5: out = y @ Wm + gsum*bm  (bf16 WMMA, f32 accum) -----
// One wave per 16x16 output tile; K=128 as 4 chained v_wmma_f32_16x16x32_bf16.
__global__ __launch_bounds__(256)
void out_gemm(const float* __restrict__ y, const __bf16* __restrict__ Bpack,
              const float* __restrict__ bm, const float* __restrict__ gsum,
              float* __restrict__ out, int S) {
  const int wave  = (blockIdx.x * blockDim.x + threadIdx.x) >> 5;
  const int lane  = threadIdx.x & 31;
  const int tilesN = DIM / 16;                      // 8
  const int tileM = wave / tilesN;
  const int tileN = wave % tilesN;
  if (tileM * 16 >= S) return;
  const int row0 = tileM * 16;
  const int col0 = tileN * 16;
  const int l15  = lane & 15;
  const int hi   = lane >> 4;                       // 0: lanes 0-15, 1: lanes 16-31

  const v16bf* Bp = reinterpret_cast<const v16bf*>(Bpack);

  v8f acc = {};
  #pragma unroll
  for (int t = 0; t < 4; ++t) {                     // K blocks of 32
    const int kbase = t * 32;
    // A (16x32 bf16, ISA layout): lane -> M=l15; VGPR v holds K pair
    //   v<4: K=2v..2v+1 (+8 if hi-half);  v>=4: K=16+2(v-4).. (+8 if hi-half)
    const float* arow = y + (size_t)(row0 + l15) * DIM + kbase + (hi ? 8 : 0);
    v16bf a;
    #pragma unroll
    for (int v = 0; v < 8; ++v) {
      int klo = (v < 4) ? (2 * v) : (16 + 2 * (v - 4));
      a[2 * v]     = (__bf16)arow[klo];
      a[2 * v + 1] = (__bf16)arow[klo + 1];
    }
    // B: pre-packed per-lane vector, one coalesced 32B load
    v16bf b = Bp[((size_t)tileN * 4 + t) * 32 + lane];

    acc = __builtin_amdgcn_wmma_f32_16x16x32_bf16(
        /*neg_a=*/false, a, /*neg_b=*/false, b,
        /*c_mod=*/(short)0, acc, /*reuse_a=*/false, /*reuse_b=*/false);
  }

  // D layout: VGPR r, lane l -> row = row0 + r + 8*hi, col = col0 + l15
  const float bmv = bm[col0 + l15];
  #pragma unroll
  for (int r = 0; r < 8; ++r) {
    int row = row0 + r + 8 * hi;
    out[(size_t)row * DIM + col0 + l15] = acc[r] + gsum[row] * bmv;
  }
}

// ---------------------------------------------------------------------------
static inline size_t align256(size_t v) { return (v + 255) & ~(size_t)255; }

extern "C" void kernel_launch(void* const* d_in, const int* in_sizes, int n_in,
                              void* d_out, int out_size, void* d_ws, size_t ws_size,
                              hipStream_t stream) {
  const float* x   = (const float*)d_in[0];
  const int*   idx = (const int*)  d_in[1];
  const float* Wg  = (const float*)d_in[2];
  const float* bg  = (const float*)d_in[3];
  const float* Wm  = (const float*)d_in[4];
  const float* bm  = (const float*)d_in[5];
  float* out = (float*)d_out;

  const int N = in_sizes[0] / DIM;   // 500000
  const int S = out_size / DIM;      // 16384

  // workspace carve-up
  char* ws = (char*)d_ws;
  size_t off = 0;
  float*    gate_raw = (float*)(ws + off);    off = align256(off + (size_t)N * 4);
  unsigned* segmax   = (unsigned*)(ws + off); off = align256(off + (size_t)S * 4);
  float*    denom    = (float*)(ws + off);    off = align256(off + (size_t)S * 4);
  float*    gsum     = (float*)(ws + off);    off = align256(off + (size_t)S * 4);
  float*    yacc     = (float*)(ws + off);    off = align256(off + (size_t)S * DIM * 4);
  __bf16*   Bpack    = (__bf16*)(ws + off);   off = align256(off + (size_t)(DIM/16) * 4 * 32 * 16 * 2);
  (void)ws_size; (void)n_in;

  // init: segmax ordinal 0 == -inf; denom/y zero
  hipMemsetAsync(segmax, 0, (size_t)S * 4, stream);
  hipMemsetAsync(denom,  0, (size_t)S * 4, stream);
  hipMemsetAsync(yacc,   0, (size_t)S * DIM * 4, stream);

  // Pass 1: gate + segment max
  {
    int blocks = (N + 7) / 8;                 // 8 waves (rows) per 256-thread block
    if (blocks > 4096) blocks = 4096;         // grid-stride
    gate_pass<<<blocks, 256, 0, stream>>>(x, idx, Wg, bg, gate_raw, segmax, N);
  }
  // Pass 2: exp + denom
  exp_pass<<<(N + 255) / 256, 256, 0, stream>>>(gate_raw, idx, segmax, denom, N);
  // Pass 3: gate mass per segment
  gsum_pass<<<(S + 255) / 256, 256, 0, stream>>>(denom, gsum, S);
  // Pass 3b: pre-pack Wm into WMMA B layout (bf16)
  pack_B<<<4, 256, 0, stream>>>(Wm, Bpack);
  // Pass 4: weighted segment sum of x (reverse chunk order for L2 reuse)
  {
    int waves  = (N + 127) / 128;
    int blocks = (waves + 7) / 8;
    segsum_pass<<<blocks, 256, 0, stream>>>(x, idx, gate_raw, denom, yacc, N, waves);
  }
  // Pass 5: WMMA GEMM epilogue
  {
    int tiles  = (S / 16) * (DIM / 16);       // one wave per tile
    int blocks = (tiles + 7) / 8;
    out_gemm<<<blocks, 256, 0, stream>>>(yacc, Bpack, bm, gsum, out, S);
  }
}
